// GraphAttention_32770600468675
// MI455X (gfx1250) — compile-verified, matrix operand order checked by reading, not run on hardware
//
#include <hip/hip_runtime.h>
#include <hip/hip_bf16.h>
#include <math.h>

// ---------------------------------------------------------------------------
// GraphAttention pipeline on gfx1250: bf16 WMMA GEMM chain.
// Block tile 256x128, 8 wave32 (4M x 2N), wave tile 64x64 (16 WMMA / K-step),
// double-buffered LDS staging (one barrier per K-step).
// ---------------------------------------------------------------------------

typedef __bf16 bf16_t;
typedef __attribute__((ext_vector_type(16))) __bf16 v16bf;
typedef __attribute__((ext_vector_type(8)))  __bf16 bf16x8;
typedef __attribute__((ext_vector_type(8)))  float  v8f;

#define BB 8
#define NN 2048
#define DF 128
#define EE 128
#define HH 4

union FragBF {
    v16bf  v;
    bf16x8 h[2];
};

// ---------------------------------------------------------------------------
// f32 -> bf16 cast (grid-stride)
// ---------------------------------------------------------------------------
__global__ __launch_bounds__(256) void cast_f32_bf16(const float* __restrict__ in,
                                                     bf16_t* __restrict__ out, long n) {
    long i = (long)blockIdx.x * 256 + threadIdx.x;
    long stride = (long)gridDim.x * 256;
    for (; i < n; i += stride) out[i] = (bf16_t)in[i];
}

// ---------------------------------------------------------------------------
// bf16 [Z][NN][EE] -> [Z][EE][NN] transpose (32x32 LDS tiles)
// ---------------------------------------------------------------------------
__global__ __launch_bounds__(256) void transpose_NE(const bf16_t* __restrict__ in,
                                                    bf16_t* __restrict__ out) {
    __shared__ bf16_t t[32][33];
    const int z  = blockIdx.z;
    const long ib = (long)z * NN * EE;
    const long ob = (long)z * EE * NN;
    const int nx = blockIdx.x * 32;          // N tile
    const int ex = blockIdx.y * 32;          // E tile
    const int cx = threadIdx.x & 31;
    const int ry = (threadIdx.x >> 5) << 2;  // 0..28 step 4
#pragma unroll
    for (int i = 0; i < 4; ++i)
        t[ry + i][cx] = in[ib + (long)(nx + ry + i) * EE + (ex + cx)];
    __syncthreads();
#pragma unroll
    for (int i = 0; i < 4; ++i)
        out[ob + (long)(ex + ry + i) * NN + (nx + cx)] = t[cx][ry + i];
}

// ---------------------------------------------------------------------------
// Generic bf16 WMMA GEMM: C[M,N] = A[M,K] * Bt[N,K]^T  (+ epilogue)
//   EP 0: store bf16
//   EP 1: + bias[col], store bf16
//   EP 2: tanh(acc * scale), store bf16
//   EP 3: relu(acc + bias[col]), store f32
// grid: (N/128, M/256, Z); block: 256 threads (8 wave32).
// Requires M%256==0, N%128==0, K%32==0 (true for all shapes used here).
// ---------------------------------------------------------------------------
template <int EP>
__global__ __launch_bounds__(256) void gemm_bf16_wmma(
    const bf16_t* __restrict__ A, long sA,
    const bf16_t* __restrict__ Bt, long sB,
    void* __restrict__ C, long sC,
    int K, int lda, int ldb, int ldc,
    const float* __restrict__ bias, float scale) {

    const int bz = blockIdx.z;
    const bf16_t* Ab = A + bz * sA;
    const bf16_t* Bb = Bt + bz * sB;

    const int tid  = threadIdx.x;
    const int lane = tid & 31;
    const int wid  = tid >> 5;       // 0..7
    const int wm   = wid & 3;        // wave M quadrant (0..3) * 64
    const int wn   = wid >> 2;       // wave N quadrant (0..1) * 64
    const int m0   = blockIdx.y * 256;
    const int n0   = blockIdx.x * 128;

    // double-buffered tiles; padded row stride (40) spreads LDS banks
    __shared__ bf16_t lA[2][256][40];
    __shared__ bf16_t lB[2][128][40];

    v8f acc[4][4] = {};

    const int sr = tid >> 2;         // staging row 0..63
    const int sc = (tid & 3) << 3;   // staging col 0,8,16,24 (16B chunks)

    auto stage = [&](int buf, int k0) {
#pragma unroll
        for (int p = 0; p < 4; ++p)
            *(bf16x8*)&lA[buf][sr + 64 * p][sc] =
                *(const bf16x8*)(Ab + (long)(m0 + sr + 64 * p) * lda + (k0 + sc));
#pragma unroll
        for (int p = 0; p < 2; ++p)
            *(bf16x8*)&lB[buf][sr + 64 * p][sc] =
                *(const bf16x8*)(Bb + (long)(n0 + sr + 64 * p) * ldb + (k0 + sc));
    };

    stage(0, 0);
    __syncthreads();

    int buf = 0;
    for (int k0 = 0; k0 < K; k0 += 32, buf ^= 1) {
        if (k0 + 32 < K) {
            stage(buf ^ 1, k0 + 32);
            if (k0 + 64 < K) {  // prefetch next-next K tile -> global_prefetch_b8
                __builtin_prefetch(Ab + (long)(m0 + sr) * lda + (k0 + 64 + sc), 0, 3);
                __builtin_prefetch(Bb + (long)(n0 + sr) * ldb + (k0 + 64 + sc), 0, 3);
            }
        }

        // WMMA fragments per the 16-bit A-layout:
        // lanes 0-15: K 0..7 (vgpr0-3), K 16..23 (vgpr4-7); lanes 16-31: +8
        const int kh = (lane >> 4) << 3;   // 0 or 8
        const int rs = lane & 15;
        FragBF a[4], b[4];
#pragma unroll
        for (int mi = 0; mi < 4; ++mi) {
            const bf16_t* p = &lA[buf][wm * 64 + mi * 16 + rs][kh];
            a[mi].h[0] = *(const bf16x8*)p;
            a[mi].h[1] = *(const bf16x8*)(p + 16);
        }
#pragma unroll
        for (int ni = 0; ni < 4; ++ni) {
            const bf16_t* p = &lB[buf][wn * 64 + ni * 16 + rs][kh];
            b[ni].h[0] = *(const bf16x8*)p;
            b[ni].h[1] = *(const bf16x8*)(p + 16);
        }
#pragma unroll
        for (int mi = 0; mi < 4; ++mi)
#pragma unroll
            for (int ni = 0; ni < 4; ++ni)
                acc[mi][ni] = __builtin_amdgcn_wmma_f32_16x16x32_bf16(
                    false, a[mi].v, false, b[ni].v, (short)0, acc[mi][ni],
                    false, false);
        __syncthreads();
    }

    // epilogue: C f32 layout — vgpr e: lanes 0-15 row=e, lanes 16-31 row=e+8
    bf16_t* Cbh = (bf16_t*)C + bz * sC;
    float*  Cbf = (float*)C + bz * sC;
    const int colb = n0 + wn * 64 + (lane & 15);
    const int rowb = m0 + wm * 64 + ((lane >> 4) << 3);
#pragma unroll
    for (int mi = 0; mi < 4; ++mi)
#pragma unroll
        for (int ni = 0; ni < 4; ++ni) {
            const int cc = colb + ni * 16;
            const int rr = rowb + mi * 16;
#pragma unroll
            for (int e = 0; e < 8; ++e) {
                float val = acc[mi][ni][e];
                if (EP == 1) val += bias[cc];
                if (EP == 2) val = tanhf(val * scale);
                if (EP == 3) val = fmaxf(val + bias[cc], 0.0f);
                const long idx = (long)(rr + e) * ldc + cc;
                if (EP == 3) Cbf[idx] = val;
                else         Cbh[idx] = (bf16_t)val;
            }
        }
}

// ---------------------------------------------------------------------------
// host-side orchestration
// ---------------------------------------------------------------------------
extern "C" void kernel_launch(void* const* d_in, const int* in_sizes, int n_in,
                              void* d_out, int out_size, void* d_ws, size_t ws_size,
                              hipStream_t stream) {
    (void)in_sizes; (void)n_in; (void)out_size; (void)ws_size;

    const float* x   = (const float*)d_in[0];
    const float* e1W = (const float*)d_in[1];
    const float* e1b = (const float*)d_in[2];
    const float* wqW = (const float*)d_in[3];
    const float* wqb = (const float*)d_in[4];
    // d_in[5], d_in[6] = wk_W / wk_b: computed in reference but unused -> skipped
    const float* wvW = (const float*)d_in[7];
    const float* wvb = (const float*)d_in[8];
    const float* e2W = (const float*)d_in[9];
    const float* e2b = (const float*)d_in[10];

    char* p = (char*)d_ws;
    auto alloc = [&](size_t bytes) {
        char* q = p;
        p += (bytes + 255) & ~(size_t)255;
        return q;
    };
    bf16_t* xb   = (bf16_t*)alloc((size_t)BB * NN * DF * 2);
    bf16_t* e1Wb = (bf16_t*)alloc((size_t)EE * NN * 2);
    bf16_t* wqWb = (bf16_t*)alloc((size_t)HH * EE * EE * 2);
    bf16_t* wvWb = (bf16_t*)alloc((size_t)HH * EE * EE * 2);
    bf16_t* e2Wb = (bf16_t*)alloc((size_t)EE * HH * EE * 2);
    bf16_t* info = (bf16_t*)alloc((size_t)BB * NN * NN * 2);   // 64 MB
    bf16_t* emb  = (bf16_t*)alloc((size_t)BB * NN * EE * 2);
    bf16_t* qb   = (bf16_t*)alloc((size_t)BB * HH * NN * EE * 2);
    bf16_t* vb   = (bf16_t*)alloc((size_t)BB * HH * NN * EE * 2);
    bf16_t* vtb  = (bf16_t*)alloc((size_t)BB * HH * EE * NN * 2);
    bf16_t* qv   = (bf16_t*)alloc((size_t)NN * NN * 2);        // reused per (b,h)
    bf16_t* attn = (bf16_t*)alloc((size_t)NN * NN * 2);        // reused per (b,h)
    bf16_t* hcat = (bf16_t*)alloc((size_t)BB * NN * HH * EE * 2);

    // casts to bf16
    cast_f32_bf16<<<1024, 256, 0, stream>>>(x,   xb,   (long)BB * NN * DF);
    cast_f32_bf16<<<256,  256, 0, stream>>>(e1W, e1Wb, (long)EE * NN);
    cast_f32_bf16<<<64,   256, 0, stream>>>(wqW, wqWb, (long)HH * EE * EE);
    cast_f32_bf16<<<64,   256, 0, stream>>>(wvW, wvWb, (long)HH * EE * EE);
    cast_f32_bf16<<<64,   256, 0, stream>>>(e2W, e2Wb, (long)EE * HH * EE);

    // info[b] = x[b] . x[b]^T        (Bt = x itself: N x DF)
    gemm_bf16_wmma<0><<<dim3(NN / 128, NN / 256, BB), 256, 0, stream>>>(
        xb, (long)NN * DF, xb, (long)NN * DF, info, (long)NN * NN,
        DF, DF, DF, NN, nullptr, 0.0f);

    // emb[b] = info[b] . e1W^T + e1b (Bt = e1W: E x N)
    gemm_bf16_wmma<1><<<dim3(EE / 128, NN / 256, BB), 256, 0, stream>>>(
        info, (long)NN * NN, e1Wb, 0, emb, (long)NN * EE,
        NN, NN, NN, EE, e1b, 0.0f);

    // q[b,h] / v[b,h] = emb[b] . W[h]^T + b[h]
    for (int h = 0; h < HH; ++h) {
        gemm_bf16_wmma<1><<<dim3(EE / 128, NN / 256, BB), 256, 0, stream>>>(
            emb, (long)NN * EE, wqWb + (long)h * EE * EE, 0,
            qb + (long)h * NN * EE, (long)HH * NN * EE,
            EE, EE, EE, EE, wqb + h * EE, 0.0f);
        gemm_bf16_wmma<1><<<dim3(EE / 128, NN / 256, BB), 256, 0, stream>>>(
            emb, (long)NN * EE, wvWb + (long)h * EE * EE, 0,
            vb + (long)h * NN * EE, (long)HH * NN * EE,
            EE, EE, EE, EE, wvb + h * EE, 0.0f);
    }

    // vT[b,h] = v[b,h]^T  (needed as Bt for h = attn . v)
    transpose_NE<<<dim3(NN / 32, EE / 32, BB * HH), 256, 0, stream>>>(vb, vtb);

    const float inv_scale = 0.08838834764831843f;  // 1/sqrt(E), E=128
    for (int bh = 0; bh < BB * HH; ++bh) {
        const int b = bh / HH, h = bh % HH;
        // qv = q . v^T   (Bt = v: N x E)
        gemm_bf16_wmma<0><<<dim3(NN / 128, NN / 256, 1), 256, 0, stream>>>(
            qb + (long)bh * NN * EE, 0, vb + (long)bh * NN * EE, 0, qv, 0,
            EE, EE, EE, NN, nullptr, 0.0f);
        // attn = tanh(qv . info / sqrt(E))  (info symmetric -> Bt = info)
        gemm_bf16_wmma<2><<<dim3(NN / 128, NN / 256, 1), 256, 0, stream>>>(
            qv, 0, info + (long)b * NN * NN, 0, attn, 0,
            NN, NN, NN, NN, nullptr, inv_scale);
        // hcat[b, :, h*E:(h+1)*E] = attn . v   (Bt = vT: E x N)
        gemm_bf16_wmma<0><<<dim3(EE / 128, NN / 256, 1), 256, 0, stream>>>(
            attn, 0, vtb + (long)bh * EE * NN, 0,
            hcat + (long)b * NN * HH * EE + (long)h * EE, 0,
            NN, NN, NN, HH * EE, nullptr, 0.0f);
    }

    // out = relu(hcat . e2W^T + e2b)  (Bt = e2W: E x H*E), f32 output
    gemm_bf16_wmma<3><<<dim3(EE / 128, NN / 256, BB), 256, 0, stream>>>(
        hcat, (long)NN * HH * EE, e2Wb, 0, d_out, (long)NN * EE,
        HH * EE, HH * EE, HH * EE, EE, e2b, 0.0f);
}
